// CRF_BiLSTM_4148938408581
// MI455X (gfx1250) — matrix operand order repaired and checked by reading, not compiled, not run on hardware
//
#include <hip/hip_runtime.h>
#include <hip/hip_bf16.h>

// ---------------------------------------------------------------------------
// CRF-BiLSTM pipeline for gfx1250 (MI455X).
//  - charcnn_kernel      : per-token conv+maxpool features (VALU)
//  - f32_to_f16 kernels  : precision staging for WMMA / packed-f16 matvec
//  - wmma_gemm4_kernel   : v_wmma_f32_16x16x32_f16, 16x64 tile per wave
//                          (A-fragment reused across 4 WMMAs per K-chunk)
//  - lstm_kernel         : 2 persistent blocks; recurrent matvec in packed
//                          f16 (v_pk_fma_f16), h-state in LDS, c in registers
//  - build_feats16       : assemble [hf|hb|x|l] in f16
//  - crf_kernel          : serial CRF forward + gold score -> scalar NLL
// ---------------------------------------------------------------------------

typedef _Float16 half16v __attribute__((ext_vector_type(16)));
typedef _Float16 half8v  __attribute__((ext_vector_type(8)));
typedef float    float8v __attribute__((ext_vector_type(8)));

#define S_LEN   8192
#define H_DIM   256
#define T_TAGS  48
#define TAG_START 46
#define TAG_END   47

// ------------------------------ conversions --------------------------------

__global__ void f32_to_f16_kernel(const float* __restrict__ src,
                                  _Float16* __restrict__ dst, int n) {
    int i = blockIdx.x * blockDim.x + threadIdx.x;
    if (i < n) dst[i] = (_Float16)src[i];
}

// dense_W is (48,1024); pad rows 48..63 with zeros -> (64,1024) f16
__global__ void densew_pad16_kernel(const float* __restrict__ src,
                                    _Float16* __restrict__ dst) {
    int i = blockIdx.x * blockDim.x + threadIdx.x;   // over 64*1024
    int r = i >> 10, c = i & 1023;
    dst[i] = (r < T_TAGS) ? (_Float16)src[r * 1024 + c] : (_Float16)0.f;
}

// ------------------------------ char CNN -----------------------------------

__global__ void charcnn_kernel(const int* __restrict__ chars,
                               const float* __restrict__ emb,
                               const float* __restrict__ cw1, const float* __restrict__ cb1,
                               const float* __restrict__ cw2, const float* __restrict__ cb2,
                               const float* __restrict__ cw3, const float* __restrict__ cb3,
                               const float* __restrict__ cw4, const float* __restrict__ cb4,
                               float* __restrict__ lout) {
    __shared__ float ce[17][16];           // (CDIM, WL)
    int s = blockIdx.x;
    int t = threadIdx.x;
    if (t < 16) {
        int ch = chars[s * 16 + t];
        #pragma unroll
        for (int c = 0; c < 17; ++c) ce[c][t] = emb[ch * 17 + c];
    }
    __syncthreads();
    int fb = t >> 6;        // filter bank 0..3 (wave-uniform: 2 waves per bank)
    int oc = t & 63;        // output channel
    float best = -3.4e38f;
    if (fb < 2) {
        const float* W = (fb ? cw2 : cw1) + oc * 34;   // (17,2)
        for (int p = 0; p < 15; ++p) {
            float v = 0.f;
            #pragma unroll
            for (int c = 0; c < 17; ++c)
                v += ce[c][p] * W[c * 2] + ce[c][p + 1] * W[c * 2 + 1];
            best = fmaxf(best, v);
        }
        best += (fb ? cb2 : cb1)[oc];
    } else {
        const float* W = ((fb == 3) ? cw4 : cw3) + oc * 51;  // (17,3)
        for (int p = 0; p < 14; ++p) {
            float v = 0.f;
            #pragma unroll
            for (int c = 0; c < 17; ++c)
                v += ce[c][p]     * W[c * 3]
                   + ce[c][p + 1] * W[c * 3 + 1]
                   + ce[c][p + 2] * W[c * 3 + 2];
            best = fmaxf(best, v);
        }
        best += ((fb == 3) ? cb4 : cb3)[oc];
    }
    lout[(size_t)s * 256 + fb * 64 + oc] = best;
}

// ------------------------------ WMMA GEMM ----------------------------------
// C[m,n] = sum_k A16[m*lda + k] * B16[n*ldb + k]  (+ bias0[n] + bias1[n])
// One wave per 16x64 output tile: 4 accumulators share one A fragment per
// K-chunk of 32 -> 4x fewer A fetches, 4 WMMAs per 10 b128 loads.

__global__ void wmma_gemm4_kernel(const _Float16* __restrict__ A,
                                  const _Float16* __restrict__ B,
                                  float* __restrict__ C,
                                  const float* __restrict__ bias0,
                                  const float* __restrict__ bias1,
                                  int M, int N, int K,
                                  int lda, int ldb, int ldo, int ncols) {
    int gwave = (blockIdx.x * blockDim.x + threadIdx.x) >> 5;
    int lane  = threadIdx.x & 31;
    int ngroups = N >> 6;                  // groups of 64 output columns
    int mtile = gwave / ngroups;
    int ngrp  = gwave % ngroups;
    if (mtile * 16 >= M) return;           // wave-uniform: EXEC stays all-ones

    int l15  = lane & 15;
    int hi   = lane >> 4;                  // 0 for lanes 0-15, 1 for lanes 16-31

    // A (16x32 f16): lane holds row M=l15; K halves {kb..kb+7, kb+16..kb+23}, kb=8*hi
    const _Float16* Aptr = A + (size_t)(mtile * 16 + l15) * lda + hi * 8;
    // B (32x16 f16): lane holds col N=l15; K halves {kb..kb+7, kb+16..kb+23}, kb=8*hi
    const _Float16* Bbase = B + (size_t)(ngrp * 64 + l15) * ldb + hi * 8;
    size_t bstep = (size_t)16 * ldb;       // next 16-column tile

    float8v acc[4] = {{}, {}, {}, {}};
    for (int kc = 0; kc < K; kc += 32) {
        half16v a;
        *(half8v*)&a       = *(const half8v*)(Aptr + kc);
        *((half8v*)&a + 1) = *(const half8v*)(Aptr + kc + 16);
        #pragma unroll
        for (int t = 0; t < 4; ++t) {
            half16v b;
            const _Float16* Bp = Bbase + (size_t)t * bstep + kc;
            *(half8v*)&b       = *(const half8v*)(Bp);
            *((half8v*)&b + 1) = *(const half8v*)(Bp + 16);
            acc[t] = __builtin_amdgcn_wmma_f32_16x16x32_f16(
                         false, a, false, b, (short)0, acc[t], false, false);
        }
    }

    // D layout: VGPR v -> row = mtile*16 + v + 8*hi; col per n-tile
    #pragma unroll
    for (int t = 0; t < 4; ++t) {
        int ncol = ngrp * 64 + t * 16 + l15;
        float badd = 0.f;
        if (ncol < ncols) {
            if (bias0) badd += bias0[ncol];
            if (bias1) badd += bias1[ncol];
        }
        #pragma unroll
        for (int v = 0; v < 8; ++v) {
            int row = mtile * 16 + v + hi * 8;
            if (row < M && ncol < ncols)
                C[(size_t)row * ldo + ncol] = acc[t][v] + badd;
        }
    }
}

// ------------------------------ LSTM recurrence ----------------------------
// blockIdx.x = direction (0 fwd, 1 bwd); 1024 threads (32 waves), persistent.
// Recurrent matvec in packed f16 (v_pk_fma_f16): 2 MACs/lane/cycle, halving
// the VALU critical path of the dominant serial phase. h in LDS (f16),
// per-channel c-state in f32 registers; W_hh (f16, 512KB/dir) L2-resident.

__device__ __forceinline__ float sigmoidf_(float x) {
    return 1.f / (1.f + __expf(-x));
}

__global__ void lstm_kernel(const float* __restrict__ pre_f,
                            const float* __restrict__ pre_b,
                            const _Float16* __restrict__ Whh16_f,
                            const _Float16* __restrict__ Whh16_b,
                            float* __restrict__ hf,
                            float* __restrict__ hb) {
    int dir = blockIdx.x;
    const float*    pre  = dir ? pre_b   : pre_f;
    const _Float16* Whh  = dir ? Whh16_b : Whh16_f;
    float*          hout = dir ? hb      : hf;

    __shared__ __align__(16) _Float16 hsh[256];
    __shared__ float gsh[1024];
    int j = threadIdx.x;
    if (j < 256) hsh[j] = (_Float16)0.f;
    float creg = 0.f;
    const half8v* Wrow = (const half8v*)(Whh + (size_t)j * 256);
    __syncthreads();

    for (int t = 0; t < S_LEN; ++t) {
        int s = dir ? (S_LEN - 1 - t) : t;
        // prefetch next step's pre-activation row (global_prefetch_b8)
        if (t + 1 < S_LEN) {
            int sn = dir ? (S_LEN - 2 - t) : (t + 1);
            __builtin_prefetch(pre + (size_t)sn * 1024 + j, 0, 3);
        }
        // packed-f16 dot: 4 parallel f16x2 accumulators, 128 v_pk_fma_f16
        half8v pacc = {};
        const half8v* hv = (const half8v*)hsh;
        #pragma unroll 8
        for (int k = 0; k < 32; ++k) {
            pacc += Wrow[k] * hv[k];
        }
        float acc = pre[(size_t)s * 1024 + j];
        #pragma unroll
        for (int e = 0; e < 8; ++e) acc += (float)pacc[e];
        gsh[j] = acc;
        __syncthreads();
        if (j < 256) {
            float ig = sigmoidf_(gsh[j]);
            float fg = sigmoidf_(gsh[256 + j]);
            float gg = tanhf(gsh[512 + j]);
            float og = sigmoidf_(gsh[768 + j]);
            creg = fg * creg + ig * gg;
            float h = og * tanhf(creg);
            hsh[j] = (_Float16)h;
            hout[(size_t)s * 256 + j] = h;
        }
        __syncthreads();
    }
}

// --------------------------- feature assembly ------------------------------

__global__ void build_feats16_kernel(const float* __restrict__ hf,
                                     const float* __restrict__ hb,
                                     const float* __restrict__ x,
                                     const float* __restrict__ l,
                                     _Float16* __restrict__ out) {
    int idx = blockIdx.x * blockDim.x + threadIdx.x;  // over S*1024
    int s = idx >> 10, c = idx & 1023;
    float v;
    if      (c < 256) v = hf[(size_t)s * 256 + c];
    else if (c < 512) v = hb[(size_t)s * 256 + (c - 256)];
    else if (c < 768) v = x [(size_t)s * 256 + (c - 512)];
    else              v = l [(size_t)s * 256 + (c - 768)];
    out[idx] = (_Float16)v;
}

// ------------------------------ CRF ----------------------------------------

__global__ void crf_kernel(const float* __restrict__ feats,
                           const float* __restrict__ trans,
                           const int* __restrict__ tags,
                           float* __restrict__ out) {
    const int T = T_TAGS;
    __shared__ float alpha[T_TAGS];
    __shared__ float red[64];
    int j = threadIdx.x;                 // 64 threads (2 waves)
    if (j < T) alpha[j] = (j == TAG_START) ? 0.f : -10000.f;
    __syncthreads();

    for (int t = 0; t < S_LEN; ++t) {
        float anew = 0.f;
        if (j < T) {
            const float* tr = trans + j * T;
            float m = -3.4e38f;
            for (int i = 0; i < T; ++i) m = fmaxf(m, alpha[i] + tr[i]);
            float sum = 0.f;
            for (int i = 0; i < T; ++i) sum += __expf(alpha[i] + tr[i] - m);
            anew = m + __logf(sum) + feats[(size_t)t * T + j];
        }
        __syncthreads();
        if (j < T) alpha[j] = anew;
        __syncthreads();
    }

    // gold path score (strided partial sums)
    float part = 0.f;
    for (int t = j; t < S_LEN; t += 64) {
        int cur  = tags[t];
        int prev = (t == 0) ? TAG_START : tags[t - 1];
        part += trans[cur * T + prev] + feats[(size_t)t * T + cur];
    }
    red[j] = part;
    __syncthreads();

    if (j == 0) {
        float gold = trans[TAG_END * T + tags[S_LEN - 1]];
        for (int i = 0; i < 64; ++i) gold += red[i];
        float m = -3.4e38f;
        for (int i = 0; i < T; ++i) m = fmaxf(m, alpha[i] + trans[TAG_END * T + i]);
        float sum = 0.f;
        for (int i = 0; i < T; ++i) sum += __expf(alpha[i] + trans[TAG_END * T + i] - m);
        float fwd = m + __logf(sum);
        out[0] = fwd - gold;
    }
}

// ------------------------------ launcher -----------------------------------

extern "C" void kernel_launch(void* const* d_in, const int* in_sizes, int n_in,
                              void* d_out, int out_size, void* d_ws, size_t ws_size,
                              hipStream_t stream) {
    (void)in_sizes; (void)n_in; (void)out_size; (void)ws_size;
    const int S = S_LEN;

    const float* sentence = (const float*)d_in[0];
    const int*   chars    = (const int*)  d_in[1];
    const int*   tags     = (const int*)  d_in[2];
    const float* emb      = (const float*)d_in[3];
    const float* trans    = (const float*)d_in[4];
    const float* W_ih_f   = (const float*)d_in[5];
    const float* W_hh_f   = (const float*)d_in[6];
    const float* b_ih_f   = (const float*)d_in[7];
    const float* b_hh_f   = (const float*)d_in[8];
    const float* W_ih_b   = (const float*)d_in[9];
    const float* W_hh_b   = (const float*)d_in[10];
    const float* b_ih_b   = (const float*)d_in[11];
    const float* b_hh_b   = (const float*)d_in[12];
    const float* dense_W  = (const float*)d_in[13];
    const float* dense_b  = (const float*)d_in[14];
    const float* cw1 = (const float*)d_in[15]; const float* cb1 = (const float*)d_in[16];
    const float* cw2 = (const float*)d_in[17]; const float* cb2 = (const float*)d_in[18];
    const float* cw3 = (const float*)d_in[19]; const float* cb3 = (const float*)d_in[20];
    const float* cw4 = (const float*)d_in[21]; const float* cb4 = (const float*)d_in[22];

    // ---- workspace carve-out (256B aligned) ----
    uint8_t* base = (uint8_t*)d_ws;
    size_t off = 0;
    auto carve = [&](size_t bytes) -> uint8_t* {
        uint8_t* p = base + off;
        off = (off + bytes + 255) & ~(size_t)255;
        return p;
    };
    _Float16* x16     = (_Float16*)carve((size_t)S * 256 * 2);
    _Float16* wihf16  = (_Float16*)carve((size_t)1024 * 256 * 2);
    _Float16* wihb16  = (_Float16*)carve((size_t)1024 * 256 * 2);
    _Float16* whhf16  = (_Float16*)carve((size_t)1024 * 256 * 2);
    _Float16* whhb16  = (_Float16*)carve((size_t)1024 * 256 * 2);
    _Float16* dW16    = (_Float16*)carve((size_t)64 * 1024 * 2);
    float*    pre_f   = (float*)   carve((size_t)S * 1024 * 4);
    float*    pre_b   = (float*)   carve((size_t)S * 1024 * 4);
    float*    hf      = (float*)   carve((size_t)S * 256 * 4);
    float*    hb      = (float*)   carve((size_t)S * 256 * 4);
    float*    lfeat   = (float*)   carve((size_t)S * 256 * 4);
    _Float16* feats16 = (_Float16*)carve((size_t)S * 1024 * 2);
    float*    featsT  = (float*)   carve((size_t)S * T_TAGS * 4);

    // ---- stage 0: precision conversions ----
    {
        int n = S * 256;
        f32_to_f16_kernel<<<(n + 255) / 256, 256, 0, stream>>>(sentence, x16, n);
        n = 1024 * 256;
        f32_to_f16_kernel<<<(n + 255) / 256, 256, 0, stream>>>(W_ih_f, wihf16, n);
        f32_to_f16_kernel<<<(n + 255) / 256, 256, 0, stream>>>(W_ih_b, wihb16, n);
        f32_to_f16_kernel<<<(n + 255) / 256, 256, 0, stream>>>(W_hh_f, whhf16, n);
        f32_to_f16_kernel<<<(n + 255) / 256, 256, 0, stream>>>(W_hh_b, whhb16, n);
        densew_pad16_kernel<<<(64 * 1024) / 256, 256, 0, stream>>>(dense_W, dW16);
    }

    // ---- stage 1: char CNN (independent of the rest) ----
    charcnn_kernel<<<S, 256, 0, stream>>>(chars, emb, cw1, cb1, cw2, cb2,
                                          cw3, cb3, cw4, cb4, lfeat);

    // ---- stage 2: input GEMMs  pre = x @ W_ih^T + b_ih + b_hh  (WMMA) ----
    {
        int waves = (S / 16) * (1024 / 64);      // 16x64 tile per wave
        int blocks = (waves * 32) / 256;
        wmma_gemm4_kernel<<<blocks, 256, 0, stream>>>(
            x16, wihf16, pre_f, b_ih_f, b_hh_f,
            S, 1024, 256, 256, 256, 1024, 1024);
        wmma_gemm4_kernel<<<blocks, 256, 0, stream>>>(
            x16, wihb16, pre_b, b_ih_b, b_hh_b,
            S, 1024, 256, 256, 256, 1024, 1024);
    }

    // ---- stage 3: serial BiLSTM recurrence (2 persistent blocks) ----
    lstm_kernel<<<2, 1024, 0, stream>>>(pre_f, pre_b, whhf16, whhb16, hf, hb);

    // ---- stage 4: assemble [hf|hb|x|l] as f16 ----
    build_feats16_kernel<<<(S * 1024) / 256, 256, 0, stream>>>(
        hf, hb, sentence, lfeat, feats16);

    // ---- stage 5: dense head  feats @ dense_W^T + b  (WMMA, N padded 48->64) ----
    {
        int waves = (S / 16) * (64 / 64);        // one 16x64 group spans all cols
        int blocks = (waves * 32) / 256;
        wmma_gemm4_kernel<<<blocks, 256, 0, stream>>>(
            feats16, dW16, featsT, dense_b, nullptr,
            S, 64, 1024, 1024, 1024, T_TAGS, T_TAGS);
    }

    // ---- stage 6: CRF forward + gold -> scalar NLL ----
    crf_kernel<<<1, 64, 0, stream>>>(featsT, trans, tags, (float*)d_out);
}
